// DFFC_63780264345960
// MI455X (gfx1250) — compile-verified, matrix operand order checked by reading, not run on hardware
//
#include <hip/hip_runtime.h>
#include <hip/hip_bf16.h>
#include <math.h>

// ---------------------------------------------------------------------------
// DFFC forward for MI455X (gfx1250). One workgroup (8 wave32) per sample.
// Every matmul is an NT GEMM (K contiguous in both operands) on
// v_wmma_f32_16x16x32_f16; fragment loads are 2x16B vector loads per operand.
// Activations live in LDS f16: time-major for the conv pipeline, token-major
// for the 128-token attention block. Weights pre-packed f32->f16 into d_ws.
// gemm_nt is force-inlined so each call site specializes; attention GEMMs
// (no row shift, full M) statically drop the boundary-guard selects.
// ---------------------------------------------------------------------------

typedef _Float16 half_t;
typedef __attribute__((ext_vector_type(16))) _Float16 v16h;
typedef __attribute__((ext_vector_type(8)))  _Float16 v8h;
typedef __attribute__((ext_vector_type(8)))  float    v8f;

#define TOK_W 104   // token-major stride [128][104]   (52 banks/row: conflict-free)
#define TA_W  136   // time-major 128ch   [80][136]
#define TB_W  264   // time-major 256ch   [80][264]
#define VT_W  136   // V^T (d-major)      [96][136]
#define S_W   136   // scores             [128][136]
#define H_W   40    // h input            [80][40]

// ---------------- param leaf order (jax sorted-key pytree flatten) ----------
enum {
  I_B2_B1 = 0, I_B2_B2, I_B2_B3,
  I_B2_BN1_BETA, I_B2_BN1_GAMMA, I_B2_BN1_MEAN, I_B2_BN1_VAR,
  I_B2_BN2_BETA, I_B2_BN2_GAMMA, I_B2_BN2_MEAN, I_B2_BN2_VAR,
  I_B2_BN3_BETA, I_B2_BN3_GAMMA, I_B2_BN3_MEAN, I_B2_BN3_VAR,
  I_B2_W1, I_B2_W2, I_B2_W3,
  I_CH_W1, I_CH_W2,
  I_C1_B1, I_C1_B2, I_C1_B3, I_C1_BN_BETA, I_C1_BN_GAMMA, I_C1_BN_MEAN, I_C1_BN_VAR,
  I_C1_W1, I_C1_W2, I_C1_W3,
  I_C2_B1, I_C2_B2, I_C2_B3, I_C2_BN_BETA, I_C2_BN_GAMMA, I_C2_BN_MEAN, I_C2_BN_VAR,
  I_C2_W1, I_C2_W2, I_C2_W3,
  I_C3_B, I_C3_BN_BETA, I_C3_BN_GAMMA, I_C3_BN_MEAN, I_C3_BN_VAR, I_C3_W,
  I_C4_B, I_C4_BN_BETA, I_C4_BN_GAMMA, I_C4_BN_MEAN, I_C4_BN_VAR, I_C4_W,
  I_C6_B, I_C6_BN_BETA, I_C6_BN_GAMMA, I_C6_BN_MEAN, I_C6_BN_VAR, I_C6_W,
  I_C8_B, I_C8_BN_BETA, I_C8_BN_GAMMA, I_C8_BN_MEAN, I_C8_BN_VAR, I_C8_W,
  I_X1_WK, I_X1_WQ, I_X1_WV, I_X1_W, I_X1_A, I_X1_B, I_X1_SCALE,
  I_X2_WK, I_X2_WQ, I_X2_WV, I_X2_W, I_X2_A, I_X2_B, I_X2_SCALE,
  I_FC_B1, I_FC_B2, I_FC_B3, I_FC_W1, I_FC_W2, I_FC_W3,
  I_SE_W1, I_SE_W2,
  I_SP_W,
  N_PARAMS
};
static_assert(N_PARAMS == 87, "param count");

struct PArr { const float* p[N_PARAMS]; };

// ---------------- packed-weight layout in d_ws (offsets in halves) ----------
constexpr size_t OFF_C1W1 = 0;                 constexpr size_t SZ_C1W1 = 3ull*128*32;
constexpr size_t OFF_C1W2 = OFF_C1W1+SZ_C1W1;  constexpr size_t SZ_C1W2 = 3ull*256*128;
constexpr size_t OFF_C1W3 = OFF_C1W2+SZ_C1W2;  constexpr size_t SZ_C1W3 = 128ull*256;
constexpr size_t OFF_C2W1 = OFF_C1W3+SZ_C1W3;  constexpr size_t SZ_C2W1 = 3ull*128*128;
constexpr size_t OFF_C2W2 = OFF_C2W1+SZ_C2W1;  constexpr size_t SZ_C2W2 = 3ull*256*128;
constexpr size_t OFF_C2W3 = OFF_C2W2+SZ_C2W2;  constexpr size_t SZ_C2W3 = 128ull*256;
constexpr size_t OFF_C3W  = OFF_C2W3+SZ_C2W3;  constexpr size_t SZ_C3W  = 128ull*128;
constexpr size_t OFF_C4W  = OFF_C3W +SZ_C3W;   constexpr size_t SZ_C4W  = 128ull*128;
constexpr size_t OFF_C6W  = OFF_C4W +SZ_C4W;   constexpr size_t SZ_C6W  = 3ull*128*128;
constexpr size_t OFF_C8W  = OFF_C6W +SZ_C6W;   constexpr size_t SZ_C8W  = 3ull*128*128;
constexpr size_t OFF_B2W1 = OFF_C8W +SZ_C8W;   constexpr size_t SZ_B2W1 = 128ull*256;
constexpr size_t OFF_B2W2 = OFF_B2W1+SZ_B2W1;  constexpr size_t SZ_B2W2 = 3ull*128*128;
constexpr size_t OFF_B2W3 = OFF_B2W2+SZ_B2W2;  constexpr size_t SZ_B2W3 = 128ull*128;
constexpr size_t OFF_XQ   = OFF_B2W3+SZ_B2W3;  constexpr size_t SZ_X    = 80ull*96;
constexpr size_t OFF_XK   = OFF_XQ + SZ_X;
constexpr size_t OFF_XV   = OFF_XK + SZ_X;
constexpr size_t OFF_XW   = OFF_XV + SZ_X;
constexpr size_t WS_HALVES = OFF_XW + SZ_X;

// ---------------- LDS arena (offsets in halves, all multiples of 8) ---------
constexpr int O_TOK0 = 0;                     // 128*104 = 13312  x_s2 (token-major)
constexpr int O_TOK1 = 13312;                 //                  x_c2
constexpr int O_TB   = 26624;                 // 80*264 = 21120   256ch time-major / X5
constexpr int O_D    = 47744;                 // overlaid region (30464)
constexpr int O_TA0  = O_D;                   // 80*136 = 10880
constexpr int O_TA1  = O_D + 10880;           // 80*136
constexpr int O_H    = O_D + 21760;           // 80*40 = 3200
constexpr int O_VT   = O_D;                   // 96*136 = 13056   (bffm phase)
constexpr int O_S    = O_D + 13056;           // 128*136 = 17408
constexpr int O_TOK2 = 78208;                 // 128*104
constexpr int O_TOK3 = 91520;                 // 128*104
constexpr int ARENA_H = 104832;               // 209,664 B

// ---------------------------------------------------------------------------
// pack kernels
// ---------------------------------------------------------------------------
__global__ void pack_convA(const float* __restrict__ src, half_t* __restrict__ dst,
                           int O, int I, int Ipad, int taps) {
  int total = taps * O * Ipad;
  for (int idx = blockIdx.x * blockDim.x + threadIdx.x; idx < total;
       idx += gridDim.x * blockDim.x) {
    int tap = idx / (O * Ipad);
    int rem = idx - tap * O * Ipad;
    int o = rem / Ipad, i = rem - o * Ipad;
    float v = (i < I) ? src[(o * I + i) * taps + tap] : 0.f;
    dst[idx] = (half_t)v;
  }
}

// [n x n] row-major -> zero-padded [rows x K] f16 (same orientation)
__global__ void pack_pad(const float* __restrict__ src, half_t* __restrict__ dst,
                         int n, int rows, int K) {
  int total = rows * K;
  for (int idx = blockIdx.x * blockDim.x + threadIdx.x; idx < total;
       idx += gridDim.x * blockDim.x) {
    int o = idx / K, d = idx - o * K;
    dst[idx] = (o < n && d < n) ? (half_t)src[o * n + d] : (half_t)0.f;
  }
}

// ---------------------------------------------------------------------------
// WMMA fragment helpers (16x16x32 f16, wave32). ISA 7.12.2 operand layout:
// lane L holds row (A) / col (B) = L&15; K-base = (L&16)?8:0;
// elems 0..7 -> K+0..7, elems 8..15 -> K+16..23.  Both operands K-contiguous
// => two aligned 16-byte vector loads per fragment.
// ---------------------------------------------------------------------------
__device__ __forceinline__ v16h zero16() {
  v16h z;
#pragma unroll
  for (int e = 0; e < 16; ++e) z[e] = (half_t)0.f;
  return z;
}

__device__ __forceinline__ v16h frag_ld(const half_t* row, int kb) {
  const v8h* p = (const v8h*)(row + kb);
  union { v16h v; v8h h[2]; } u;
  u.h[0] = p[0];
  u.h[1] = p[2];          // row + kb + 16
  return u.v;
}

// A fragment; when `noguard` is a compile-time constant true (attention GEMMs)
// the clamp+select vanish entirely.
__device__ __forceinline__ v16h load_A(const half_t* A, int As, int m0, int k0,
                                       int shift, int Mact, int lane, bool noguard) {
  int m = m0 + (lane & 15) + shift;
  int kb = k0 + ((lane & 16) ? 8 : 0);
  if (noguard) return frag_ld(A + (size_t)m * As, kb);
  bool ok = (m >= 0) && (m < Mact);
  int mc = ok ? m : 0;
  v16h a = frag_ld(A + (size_t)mc * As, kb);
  if (!ok) a = zero16();
  return a;
}

__device__ __forceinline__ v16h load_B(const half_t* Bt, int Bs, int n0, int k0,
                                       int lane) {
  int n = n0 + (lane & 15);
  int kb = k0 + ((lane & 16) ? 8 : 0);
  return frag_ld(Bt + (size_t)n * Bs, kb);
}

struct EpiP {
  const float* bias;                 // indexed by n
  const float *g, *bb, *mu, *vv;     // BN by n (g==null -> none)
  int relu;
  float mul;                         // pre-scale (scores)
  int mode;                          // 0: D[m][n]; 1: D[n][m]; 2: D[n][m]=ca*R+cb*relu(v)+res
  float ca, cb;
  const half_t* R; const half_t* res; // token-major stride TOK_W, indexed [m][n]
};
__device__ __forceinline__ EpiP mkepi() {
  EpiP e; e.bias = nullptr; e.g = nullptr; e.bb = nullptr; e.mu = nullptr;
  e.vv = nullptr; e.relu = 0; e.mul = 1.f; e.mode = 0; e.ca = 0.f; e.cb = 0.f;
  e.R = nullptr; e.res = nullptr; return e;
}

// NT GEMM: C[m][n] = epi( sum_tap sum_k A[m+tap-ctr][k] * Bt_tap[n][k] )
// Force-inlined: every call site has constant dims/strides/epilogue config,
// so guards, null checks and mode switches fold away per site.
__device__ __forceinline__ void gemm_nt(const half_t* A, int As, int Mact,
                                        const half_t* Bt, int Bs,
                                        half_t* D, int Ds,
                                        int M, int N, int Kp,
                                        int taps, int tapStride, int ctr,
                                        const EpiP& ep, int tid) {
  const int lane = tid & 31, wave = tid >> 5;
  const int TN = N >> 4, TT = (M >> 4) * TN;
  const bool noguard = (taps == 1) && (ctr == 0) && (Mact >= M);   // folds per site
  for (int t = wave; t < TT; t += 8) {
    const int m0 = (t / TN) << 4, n0 = (t % TN) << 4;
    v8f acc = {0.f, 0.f, 0.f, 0.f, 0.f, 0.f, 0.f, 0.f};
#pragma unroll 1
    for (int tap = 0; tap < taps; ++tap) {
      const half_t* Bp = Bt + (size_t)tap * tapStride;
#pragma unroll
      for (int k0 = 0; k0 < Kp; k0 += 32) {
        v16h a = load_A(A, As, m0, k0, tap - ctr, Mact, lane, noguard);
        v16h b = load_B(Bp, Bs, n0, k0, lane);
        acc = __builtin_amdgcn_wmma_f32_16x16x32_f16(false, a, false, b,
                                                     (short)0, acc, false, false);
      }
    }
    const int n = n0 + (lane & 15);
    const int mo = (lane & 16) ? 8 : 0;
    float bias = ep.bias ? ep.bias[n] : 0.f;
    float bs = 1.f, bo = 0.f;
    if (ep.g) { bs = ep.g[n] * rsqrtf(ep.vv[n] + 1e-5f); bo = ep.bb[n] - ep.mu[n] * bs; }
#pragma unroll
    for (int r = 0; r < 8; ++r) {
      int m = m0 + mo + r;
      float v = acc[r] * ep.mul + bias;
      if (ep.g) v = v * bs + bo;
      if (ep.relu) v = fmaxf(v, 0.f);
      if (ep.mode == 2)
        v = ep.ca * (float)ep.R[m * TOK_W + n] + ep.cb * v + (float)ep.res[m * TOK_W + n];
      if (ep.mode) D[(size_t)n * Ds + m] = (half_t)v;
      else         D[(size_t)m * Ds + n] = (half_t)v;
    }
  }
}

// ---------------------------------------------------------------------------
// cifm for the 128-token bffm: fdst(transposed) = ca*R + cb*relu((V-R) W^T) + res
// ---------------------------------------------------------------------------
__device__ __forceinline__ void cifm128(
    const half_t* q_in, const half_t* kv_in, const half_t* res,
    half_t* Qb, half_t* Kb, half_t* Vt, half_t* Sb,
    half_t* fdst,            // X5 column base, stride TB_W
    const half_t* ws, float ca, float cb, float smul, int tid) {
  EpiP e0 = mkepi();
  // Q[c][o], K[c][o] token-major; V written d-major via transposed epilogue
  gemm_nt(q_in,  TOK_W, 128, ws + OFF_XQ, 96, Qb, TOK_W, 128, 80, 96, 1, 0, 0, e0, tid);
  gemm_nt(kv_in, TOK_W, 128, ws + OFF_XK, 96, Kb, TOK_W, 128, 80, 96, 1, 0, 0, e0, tid);
  EpiP eT = mkepi(); eT.mode = 1;
  gemm_nt(kv_in, TOK_W, 128, ws + OFF_XV, 96, Vt, VT_W, 128, 80, 96, 1, 0, 0, eT, tid);
  __syncthreads();
  // scores = (Q K^T) * scale/sqrt(70)   (NT: both token-major, K-contiguous)
  EpiP es = mkepi(); es.mul = smul;
  gemm_nt(Qb, TOK_W, 128, Kb, TOK_W, Sb, S_W, 128, 128, 96, 1, 0, 0, es, tid);
  __syncthreads();
  // softmax rows
  if (tid < 128) {
    half_t* row = Sb + tid * S_W;
    float mx = -1e30f;
    for (int j = 0; j < 128; ++j) mx = fmaxf(mx, (float)row[j]);
    float sum = 0.f;
    for (int j = 0; j < 128; ++j) {
      float v = __expf((float)row[j] - mx);
      sum += v; row[j] = (half_t)v;
    }
    float inv = 1.f / sum;
    for (int j = 0; j < 128; ++j) row[j] = (half_t)((float)row[j] * inv);
  }
  __syncthreads();
  // R = relu(softmax @ V): A = S (K = tokens), Bt = V^T (d-major) -> overwrite Qb
  EpiP er = mkepi(); er.relu = 1;
  gemm_nt(Sb, S_W, 128, Vt, VT_W, Qb, TOK_W, 128, 80, 128, 1, 0, 0, er, tid);
  __syncthreads();
  // V - R -> Kb (token-major); V read transposed from Vt
  for (int i = tid; i < 128 * 80; i += 256) {
    int c = i / 80, j = i - c * 80;
    Kb[c * TOK_W + j] = (half_t)((float)Vt[j * VT_W + c] - (float)Qb[c * TOK_W + j]);
  }
  __syncthreads();
  // f = ca*R + cb*relu((V-R) W^T) + res, stored transposed into time-major X5
  EpiP ec = mkepi(); ec.relu = 1; ec.mode = 2; ec.ca = ca; ec.cb = cb;
  ec.R = Qb; ec.res = res;
  gemm_nt(Kb, TOK_W, 128, ws + OFF_XW, 96, fdst, TB_W, 128, 80, 96, 1, 0, 0, ec, tid);
}

// ---------------------------------------------------------------------------
// main kernel: one block per sample
// ---------------------------------------------------------------------------
__global__ __launch_bounds__(256, 1)
void dffc_main(const float* __restrict__ xg, float* __restrict__ outp,
               const half_t* __restrict__ ws, PArr P) {
  __shared__ __align__(16) half_t arena[ARENA_H];
  __shared__ float fscr[1024];

  half_t* TOK0 = arena + O_TOK0;
  half_t* TOK1 = arena + O_TOK1;
  half_t* TB   = arena + O_TB;
  half_t* TA0  = arena + O_TA0;
  half_t* TA1  = arena + O_TA1;
  half_t* Hh   = arena + O_H;
  half_t* VT   = arena + O_VT;
  half_t* Sb   = arena + O_S;
  half_t* TOK2 = arena + O_TOK2;
  half_t* TOK3 = arena + O_TOK3;

  const int tid = threadIdx.x;
  const int b   = blockIdx.x;

  // ---- zero arena (supplies all zero padding); load sample ----------------
  {
    uint4* az = (uint4*)arena;
    for (int i = tid; i < ARENA_H / 8; i += 256) az[i] = uint4{0u, 0u, 0u, 0u};
  }
  if (tid < 140) {
    float v = xg[b * 140 + tid];
    fscr[(tid < 70) ? tid : (80 + tid - 70)] = v;   // xa @0, xb @80
  }
  __syncthreads();

  // ---- bffm #1 (seq len 1: softmax==1) -> h time-major in Hh --------------
  {
    const float* Wv = P.p[I_X1_WV];
    if (tid < 140) {
      int off = (tid < 70) ? 0 : 80;
      int o   = (tid < 70) ? tid : (tid - 70);
      const float* xin = fscr + off;
      float s = 0.f;
      for (int d = 0; d < 70; ++d) s += xin[d] * Wv[o * 70 + d];
      fscr[160 + off + o] = s;              // V
      fscr[320 + off + o] = fminf(s, 0.f);  // V - relu(V)
    }
    __syncthreads();
    const float* Wc = P.p[I_X1_W];
    const float a1 = P.p[I_X1_A][0], b1 = P.p[I_X1_B][0];
    if (tid < 140) {
      int off = (tid < 70) ? 0 : 80;
      int o   = (tid < 70) ? tid : (tid - 70);
      float s = 0.f;
      const float* m = fscr + 320 + off;
      for (int d = 0; d < 70; ++d) s += m[d] * Wc[o * 70 + d];
      float cc  = fmaxf(s, 0.f);
      float r   = fmaxf(fscr[160 + off + o], 0.f);
      float val = a1 * r + b1 * cc + fscr[off + o];
      Hh[o * H_W + ((tid < 70) ? 0 : 1)] = (half_t)val;   // time-major [t][ch]
    }
  }
  __syncthreads();

  // ---- seq3 #1 (time-major): h -> x1 in TA1 -------------------------------
  { EpiP e = mkepi(); e.bias = P.p[I_C1_B1];
    gemm_nt(Hh, H_W, 70, ws + OFF_C1W1, 32, TA0, TA_W, 80, 128, 32, 3, 128*32, 1, e, tid); }
  __syncthreads();
  { EpiP e = mkepi(); e.bias = P.p[I_C1_B2];
    gemm_nt(TA0, TA_W, 70, ws + OFF_C1W2, 128, TB, TB_W, 80, 256, 128, 3, 256*128, 1, e, tid); }
  __syncthreads();
  { EpiP e = mkepi(); e.bias = P.p[I_C1_B3];
    e.g = P.p[I_C1_BN_GAMMA]; e.bb = P.p[I_C1_BN_BETA];
    e.mu = P.p[I_C1_BN_MEAN]; e.vv = P.p[I_C1_BN_VAR]; e.relu = 1;
    gemm_nt(TB, TB_W, 70, ws + OFF_C1W3, 256, TA1, TA_W, 80, 128, 256, 1, 0, 0, e, tid); }
  __syncthreads();

  // ---- seq3 #2: x1 -> x2 in TA1 -------------------------------------------
  { EpiP e = mkepi(); e.bias = P.p[I_C2_B1];
    gemm_nt(TA1, TA_W, 70, ws + OFF_C2W1, 128, TA0, TA_W, 80, 128, 128, 3, 128*128, 1, e, tid); }
  __syncthreads();
  { EpiP e = mkepi(); e.bias = P.p[I_C2_B2];
    gemm_nt(TA0, TA_W, 70, ws + OFF_C2W2, 128, TB, TB_W, 80, 256, 128, 3, 256*128, 1, e, tid); }
  __syncthreads();
  { EpiP e = mkepi(); e.bias = P.p[I_C2_B3];
    e.g = P.p[I_C2_BN_GAMMA]; e.bb = P.p[I_C2_BN_BETA];
    e.mu = P.p[I_C2_BN_MEAN]; e.vv = P.p[I_C2_BN_VAR]; e.relu = 1;
    gemm_nt(TB, TB_W, 70, ws + OFF_C2W3, 256, TA1, TA_W, 80, 128, 256, 1, 0, 0, e, tid); }
  __syncthreads();

  // ---- SE attention on x2 (TA1, time-major), in place ---------------------
  if (tid < 128) {
    float s = 0.f;
    for (int t = 0; t < 70; ++t) s += (float)TA1[t * TA_W + tid];
    fscr[tid] = s * (1.f / 70.f);
  }
  __syncthreads();
  if (tid < 8) {
    const float* w1 = P.p[I_SE_W1];
    float s = 0.f;
    for (int c = 0; c < 128; ++c) s += fscr[c] * w1[tid * 128 + c];
    fscr[128 + tid] = fmaxf(s, 0.f);
  }
  __syncthreads();
  if (tid < 128) {
    const float* w2 = P.p[I_SE_W2];
    float s = 0.f;
    for (int j = 0; j < 8; ++j) s += fscr[128 + j] * w2[tid * 8 + j];
    fscr[256 + tid] = 1.f / (1.f + __expf(-s));
  }
  __syncthreads();
  for (int i = tid; i < 70 * 128; i += 256) {
    int t = i >> 7, c = i & 127;
    TA1[t * TA_W + c] = (half_t)((float)TA1[t * TA_W + c] * fscr[256 + c]);
  }
  __syncthreads();

  // ---- x3 = cbr(conv3) -> TA0 ; x4 = cbr(conv4) -> TB ---------------------
  { EpiP e = mkepi(); e.bias = P.p[I_C3_B];
    e.g = P.p[I_C3_BN_GAMMA]; e.bb = P.p[I_C3_BN_BETA];
    e.mu = P.p[I_C3_BN_MEAN]; e.vv = P.p[I_C3_BN_VAR]; e.relu = 1;
    gemm_nt(TA1, TA_W, 70, ws + OFF_C3W, 128, TA0, TA_W, 80, 128, 128, 1, 0, 0, e, tid); }
  __syncthreads();
  { EpiP e = mkepi(); e.bias = P.p[I_C4_B];
    e.g = P.p[I_C4_BN_GAMMA]; e.bb = P.p[I_C4_BN_BETA];
    e.mu = P.p[I_C4_BN_MEAN]; e.vv = P.p[I_C4_BN_VAR]; e.relu = 1;
    gemm_nt(TA1, TA_W, 70, ws + OFF_C4W, 128, TB, TB_W, 80, 128, 128, 1, 0, 0, e, tid); }
  __syncthreads();

  // ---- spatial attention on x3 (TA0), in place ----------------------------
  if (tid < 70) {
    float s = 0.f, m = -1e30f;
    const half_t* row = TA0 + tid * TA_W;
    for (int c = 0; c < 128; ++c) {
      float v = (float)row[c];
      s += v; m = fmaxf(m, v);
    }
    fscr[tid] = s * (1.f / 128.f); fscr[80 + tid] = m;
  }
  __syncthreads();
  if (tid < 70) {
    const float* w = P.p[I_SP_W];          // [1][2][7]
    float s = 0.f;
    for (int j = 0; j < 7; ++j) {
      int t = tid + j - 3;
      if (t >= 0 && t < 70) s += fscr[t] * w[j] + fscr[80 + t] * w[7 + j];
    }
    fscr[160 + tid] = 1.f / (1.f + __expf(-s));
  }
  __syncthreads();
  for (int i = tid; i < 70 * 128; i += 256) {
    int t = i >> 7, c = i & 127;
    TA0[t * TA_W + c] = (half_t)((float)TA0[t * TA_W + c] * fscr[160 + t]);
  }
  __syncthreads();

  // ---- channel attention on x4 (TB), in place -----------------------------
  if (tid < 128) {
    float s = 0.f, m = -1e30f;
    for (int t = 0; t < 70; ++t) {
      float v = (float)TB[t * TB_W + tid];
      s += v; m = fmaxf(m, v);
    }
    fscr[tid] = s * (1.f / 70.f); fscr[128 + tid] = m;
  }
  __syncthreads();
  if (tid < 16) {
    const float* w1 = P.p[I_CH_W1];        // [8][128]
    int j = tid & 7;
    const float* src = fscr + ((tid >> 3) ? 128 : 0);
    float s = 0.f;
    for (int c = 0; c < 128; ++c) s += src[c] * w1[j * 128 + c];
    fscr[256 + tid] = fmaxf(s, 0.f);
  }
  __syncthreads();
  if (tid < 128) {
    const float* w2 = P.p[I_CH_W2];        // [128][8]; final conv is linear
    float s = 0.f;
    for (int j = 0; j < 8; ++j) s += (fscr[256 + j] + fscr[264 + j]) * w2[tid * 8 + j];
    fscr[288 + tid] = 1.f / (1.f + __expf(-s));
  }
  __syncthreads();
  for (int i = tid; i < 70 * 128; i += 256) {
    int t = i >> 7, c = i & 127;
    TB[t * TB_W + c] = (half_t)((float)TB[t * TB_W + c] * fscr[288 + c]);
  }
  __syncthreads();

  // ---- x_s2 = cbr(conv6, x_s) -> TOK0 (token-major, transposed store) -----
  { EpiP e = mkepi(); e.bias = P.p[I_C6_B];
    e.g = P.p[I_C6_BN_GAMMA]; e.bb = P.p[I_C6_BN_BETA];
    e.mu = P.p[I_C6_BN_MEAN]; e.vv = P.p[I_C6_BN_VAR]; e.relu = 1; e.mode = 1;
    gemm_nt(TA0, TA_W, 70, ws + OFF_C6W, 128, TOK0, TOK_W, 80, 128, 128, 3, 128*128, 1, e, tid); }
  __syncthreads();
  // ---- x_c2 = cbr(conv8, x_c) -> TOK1 -------------------------------------
  { EpiP e = mkepi(); e.bias = P.p[I_C8_B];
    e.g = P.p[I_C8_BN_GAMMA]; e.bb = P.p[I_C8_BN_BETA];
    e.mu = P.p[I_C8_BN_MEAN]; e.vv = P.p[I_C8_BN_VAR]; e.relu = 1; e.mode = 1;
    gemm_nt(TB, TB_W, 70, ws + OFF_C8W, 128, TOK1, TOK_W, 80, 128, 128, 3, 128*128, 1, e, tid); }
  __syncthreads();

  // ---- bffm #2 on x = TOK0, y = TOK1 --------------------------------------
  const float ca2  = P.p[I_X2_A][0];
  const float cb2  = P.p[I_X2_B][0];
  const float smul = P.p[I_X2_SCALE][0] * 0.11952286093343936f;  // 1/sqrt(70)
  // f1 = cifm(Qy, Kx, Vx) + x -> X5 cols 0..127 (time-major)
  cifm128(TOK1, TOK0, TOK0, TOK2, TOK3, VT, Sb, TB, ws, ca2, cb2, smul, tid);
  __syncthreads();
  // f2 = cifm(Qx, Ky, Vy) + y -> X5 cols 128..255
  cifm128(TOK0, TOK1, TOK1, TOK2, TOK3, VT, Sb, TB + 128, ws, ca2, cb2, smul, tid);
  __syncthreads();

  // ---- bott2 on X5 = TB [80][256] -----------------------------------------
  { EpiP e = mkepi(); e.bias = P.p[I_B2_B1];
    e.g = P.p[I_B2_BN1_GAMMA]; e.bb = P.p[I_B2_BN1_BETA];
    e.mu = P.p[I_B2_BN1_MEAN]; e.vv = P.p[I_B2_BN1_VAR]; e.relu = 1;
    gemm_nt(TB, TB_W, 70, ws + OFF_B2W1, 256, TA0, TA_W, 80, 128, 256, 1, 0, 0, e, tid); }
  __syncthreads();
  { EpiP e = mkepi(); e.bias = P.p[I_B2_B2];
    e.g = P.p[I_B2_BN2_GAMMA]; e.bb = P.p[I_B2_BN2_BETA];
    e.mu = P.p[I_B2_BN2_MEAN]; e.vv = P.p[I_B2_BN2_VAR]; e.relu = 1;
    gemm_nt(TA0, TA_W, 70, ws + OFF_B2W2, 128, TA1, TA_W, 80, 128, 128, 3, 128*128, 1, e, tid); }
  __syncthreads();
  { EpiP e = mkepi(); e.bias = P.p[I_B2_B3];
    e.g = P.p[I_B2_BN3_GAMMA]; e.bb = P.p[I_B2_BN3_BETA];
    e.mu = P.p[I_B2_BN3_MEAN]; e.vv = P.p[I_B2_BN3_VAR]; e.relu = 1;
    gemm_nt(TA1, TA_W, 70, ws + OFF_B2W3, 128, TA0, TA_W, 80, 128, 128, 1, 0, 0, e, tid); }
  __syncthreads();

  // ---- global average pool + FC head --------------------------------------
  if (tid < 128) {
    float s = 0.f;
    for (int t = 0; t < 70; ++t) s += (float)TA0[t * TA_W + tid];
    fscr[tid] = s * (1.f / 70.f);
  }
  __syncthreads();
  if (tid < 64) {
    const float* w = P.p[I_FC_W1];
    float s = P.p[I_FC_B1][tid];
    for (int c = 0; c < 128; ++c) s += fscr[c] * w[tid * 128 + c];
    fscr[128 + tid] = s;
  }
  __syncthreads();
  if (tid < 32) {
    const float* w = P.p[I_FC_W2];
    float s = P.p[I_FC_B2][tid];
    for (int j = 0; j < 64; ++j) s += fscr[128 + j] * w[tid * 64 + j];
    fscr[192 + tid] = s;
  }
  __syncthreads();
  if (tid < 5) {
    const float* w = P.p[I_FC_W3];
    float s = P.p[I_FC_B3][tid];
    for (int j = 0; j < 32; ++j) s += fscr[192 + j] * w[tid * 32 + j];
    outp[b * 5 + tid] = s;
  }
}

// ---------------------------------------------------------------------------
// host
// ---------------------------------------------------------------------------
extern "C" void kernel_launch(void* const* d_in, const int* in_sizes, int n_in,
                              void* d_out, int out_size, void* d_ws, size_t ws_size,
                              hipStream_t stream) {
  int ix = -1;
  for (int i = 0; i < n_in; ++i)
    if (in_sizes[i] == 1024 * 140) { ix = i; break; }
  if (ix < 0) ix = 0;

  PArr P;
  int j = 0;
  for (int i = 0; i < n_in && j < N_PARAMS; ++i) {
    if (i == ix) continue;
    P.p[j++] = (const float*)d_in[i];
  }
  for (; j < N_PARAMS; ++j) P.p[j] = (const float*)d_in[ix];

  const float* xg = (const float*)d_in[ix];
  const int Bn = in_sizes[ix] / 140;
  half_t* ws = (half_t*)d_ws;
  (void)ws_size; (void)out_size;

  auto packA = [&](int pi, size_t off, int O, int I, int Ipad, int taps) {
    int total = taps * O * Ipad;
    pack_convA<<<(total + 255) / 256, 256, 0, stream>>>(P.p[pi], ws + off, O, I, Ipad, taps);
  };
  packA(I_C1_W1, OFF_C1W1, 128,   2,  32, 3);
  packA(I_C1_W2, OFF_C1W2, 256, 128, 128, 3);
  packA(I_C1_W3, OFF_C1W3, 128, 256, 256, 1);
  packA(I_C2_W1, OFF_C2W1, 128, 128, 128, 3);
  packA(I_C2_W2, OFF_C2W2, 256, 128, 128, 3);
  packA(I_C2_W3, OFF_C2W3, 128, 256, 256, 1);
  packA(I_C3_W,  OFF_C3W,  128, 128, 128, 1);
  packA(I_C4_W,  OFF_C4W,  128, 128, 128, 1);
  packA(I_C6_W,  OFF_C6W,  128, 128, 128, 3);
  packA(I_C8_W,  OFF_C8W,  128, 128, 128, 3);
  packA(I_B2_W1, OFF_B2W1, 128, 256, 256, 1);
  packA(I_B2_W2, OFF_B2W2, 128, 128, 128, 3);
  packA(I_B2_W3, OFF_B2W3, 128, 128, 128, 1);

  auto packP = [&](int pi, size_t off) {
    int total = 80 * 96;
    pack_pad<<<(total + 255) / 256, 256, 0, stream>>>(P.p[pi], ws + off, 70, 80, 96);
  };
  packP(I_X2_WQ, OFF_XQ);
  packP(I_X2_WK, OFF_XK);
  packP(I_X2_WV, OFF_XV);
  packP(I_X2_W,  OFF_XW);

  dffc_main<<<Bn, 256, 0, stream>>>(xg, (float*)d_out, ws, P);
}
